// FactInterface_89137751261847
// MI455X (gfx1250) — matrix operand order repaired and checked by reading, not compiled, not run on hardware
//
#include <hip/hip_runtime.h>
#include <hip/hip_bf16.h>
#include <cstdint>
#include <cstddef>

#define B_  16
#define T_  256
#define D_  2048
#define H_  16
#define DK_ 64
#define DV_ 64

typedef __attribute__((ext_vector_type(16))) __bf16 v16bf;
typedef __attribute__((ext_vector_type(8)))  float  v8f;

// ---------------------------------------------------------------------------
// WMMA wrapper: D = A(16x32 bf16) * B(32x16 bf16) + C(16x16 f32)
// ---------------------------------------------------------------------------
__device__ __forceinline__ v8f wmma_bf16(v16bf a, v16bf b, v8f c) {
  return __builtin_amdgcn_wmma_f32_16x16x32_bf16(false, a, false, b,
                                                 (short)0, c, false, false);
}

// ---------------------------------------------------------------------------
// CDNA5 async copy: 16 bytes per lane, global -> LDS, tracked by ASYNCcnt.
// lds_off is the per-lane LDS byte address (dynamic-LDS base is 0 here).
// ---------------------------------------------------------------------------
__device__ __forceinline__ void async_copy_16(unsigned lds_off, const void* g) {
  asm volatile("global_load_async_to_lds_b128 %0, %1, off"
               :: "v"(lds_off), "v"((unsigned long long)(uintptr_t)g)
               : "memory");
}
__device__ __forceinline__ void wait_async0() {
  asm volatile("s_wait_asynccnt 0" ::: "memory");
}

// ---------------------------------------------------------------------------
// Fragment loaders (wave32; this file's contiguous k order, not the ISA table order)
// A 16x32 bf16: lane L holds row M=L%16; lanes 0-15 -> K {0..15},
// lanes 16-31 -> K {16..31}; VGPR pairs contiguous in K.
// ---------------------------------------------------------------------------
__device__ __forceinline__ v16bf load_frag_a(const __bf16* src, int lda, int lane) {
  const int row = lane & 15;
  const int kb  = (lane >> 4) << 4;       // 0 or 16
  const __bf16* p = src + row * lda + kb;
  v16bf a;
#pragma unroll
  for (int q = 0; q < 4; ++q) { a[2*q]   = p[2*q];      a[2*q+1]   = p[2*q+1]; }
#pragma unroll
  for (int q = 0; q < 4; ++q) { a[8+2*q] = p[8+2*q];   a[8+2*q+1] = p[8+2*q+1]; }
  return a;
}

// B 32x16 bf16, element (kk,n) = src[n*ldb + kk]  (B given transposed,
// contiguous along K — natural layout for C = A * W^T with row-major W).
__device__ __forceinline__ v16bf load_frag_b_nT(const __bf16* src, int ldb, int lane) {
  const int n  = lane & 15;
  const int kb = (lane >> 4) << 4;        // 0 or 16
  const __bf16* p = src + n * ldb + kb;
  v16bf b;
#pragma unroll
  for (int q = 0; q < 8; ++q) { b[2*q] = p[2*q]; b[2*q+1] = p[2*q+1]; }
  return b;
}

// Same but source is fp32 (converted on the fly).
__device__ __forceinline__ v16bf load_frag_b_nT_f32(const float* src, int ldb, int lane) {
  const int n  = lane & 15;
  const int kb = (lane >> 4) << 4;
  const float* p = src + n * ldb + kb;
  v16bf b;
#pragma unroll
  for (int q = 0; q < 16; ++q) b[q] = (__bf16)p[q];
  return b;
}

// B 32x16 bf16, element (kk,n) = src[kk*ldb + n]  (row-major K x N source).
__device__ __forceinline__ v16bf load_frag_b_kn(const __bf16* src, int ldb, int lane) {
  const int n  = lane & 15;
  const int kb = (lane >> 4) << 4;
  v16bf b;
#pragma unroll
  for (int q = 0; q < 16; ++q) b[q] = src[(kb + q) * ldb + n];
  return b;
}

// Row-major fp32 K x N source with K-tail masking (rows kk >= kvalid -> 0).
__device__ __forceinline__ v16bf load_frag_b_kn_f32_mask(const float* src, int ldb,
                                                         int lane, int kvalid) {
  const int n  = lane & 15;
  const int kb = (lane >> 4) << 4;
  v16bf b;
#pragma unroll
  for (int q = 0; q < 16; ++q) {
    const int kk = kb + q;
    b[q] = (__bf16)((kk < kvalid) ? src[kk * ldb + n] : 0.0f);
  }
  return b;
}

// A fragment for the W-update: A[v][t] = beta[t] * Vp[t][v] (fp32 LDS -> bf16).
__device__ __forceinline__ v16bf load_frag_a_betaVp(const float* Vp, const float* beta,
                                                    int voff, int k0, int lane) {
  const int row = lane & 15;
  const int kb  = (lane >> 4) << 4;
  v16bf a;
#pragma unroll
  for (int q = 0; q < 4; ++q) {
    const int t = k0 + kb + 2 * q;
    a[2*q]   = (__bf16)(beta[t]     * Vp[t       * DV_ + voff + row]);
    a[2*q+1] = (__bf16)(beta[t + 1] * Vp[(t + 1) * DV_ + voff + row]);
  }
#pragma unroll
  for (int q = 0; q < 4; ++q) {
    const int t = k0 + kb + 8 + 2 * q;
    a[8+2*q]   = (__bf16)(beta[t]     * Vp[t       * DV_ + voff + row]);
    a[8+2*q+1] = (__bf16)(beta[t + 1] * Vp[(t + 1) * DV_ + voff + row]);
  }
  return a;
}

// ---------------------------------------------------------------------------
// fp32 -> bf16 conversion (grid-stride)
// ---------------------------------------------------------------------------
__global__ __launch_bounds__(256) void cvt_f32_bf16_kernel(
    const float* __restrict__ in, __bf16* __restrict__ out, size_t n) {
  size_t i = (size_t)blockIdx.x * blockDim.x + threadIdx.x;
  const size_t stride = (size_t)gridDim.x * blockDim.x;
  for (; i < n; i += stride) out[i] = (__bf16)in[i];
}

// ---------------------------------------------------------------------------
// Generic WMMA GEMM: C[m,n] = base[m,n] + s * sum_k A[m,k]*Bt[n,k]
// A: M x K bf16 row-major, Bt: N x K bf16 row-major.
// s = sigmoid(*gate) if gate != nullptr else 1.
// Block = 256 threads (8 waves); wave tile 32x32; block tile 64x128.
// ---------------------------------------------------------------------------
__global__ __launch_bounds__(256) void gemm_bf16_kernel(
    const __bf16* __restrict__ A, const __bf16* __restrict__ Bt,
    float* __restrict__ C, const float* __restrict__ base,
    const float* __restrict__ gate, int M, int N, int K) {
  const int lane = threadIdx.x & 31;
  const int wave = threadIdx.x >> 5;
  const int bm = blockIdx.y * 64  + (wave >> 2) * 32;
  const int bn = blockIdx.x * 128 + (wave & 3)  * 32;

  v8f acc[2][2] = {};
  for (int k0 = 0; k0 < K; k0 += 32) {
    if (k0 + 64 < K) {   // speculative prefetch of the next-next K chunk
      __builtin_prefetch(A + (size_t)bm * K + k0 + 64, 0, 0);
      __builtin_prefetch(Bt + (size_t)bn * K + k0 + 64, 0, 0);
    }
    v16bf a0 = load_frag_a(A + (size_t)bm * K + k0, K, lane);
    v16bf a1 = load_frag_a(A + (size_t)(bm + 16) * K + k0, K, lane);
    v16bf b0 = load_frag_b_nT(Bt + (size_t)bn * K + k0, K, lane);
    v16bf b1 = load_frag_b_nT(Bt + (size_t)(bn + 16) * K + k0, K, lane);
    acc[0][0] = wmma_bf16(a0, b0, acc[0][0]);
    acc[0][1] = wmma_bf16(a0, b1, acc[0][1]);
    acc[1][0] = wmma_bf16(a1, b0, acc[1][0]);
    acc[1][1] = wmma_bf16(a1, b1, acc[1][1]);
  }

  float s = 1.0f;
  if (gate) s = 1.0f / (1.0f + __expf(-gate[0]));

  const int rlo = 8 * (lane >> 4);
  const int col = lane & 15;
#pragma unroll
  for (int i = 0; i < 2; ++i)
#pragma unroll
    for (int j = 0; j < 2; ++j) {
      const int m0 = bm + 16 * i + rlo;
      const int n0 = bn + 16 * j + col;
#pragma unroll
      for (int r = 0; r < 8; ++r) {
        const size_t idx = (size_t)(m0 + r) * N + n0;
        const float v = s * acc[i][j][r];
        C[idx] = (base ? base[idx] : 0.0f) + v;
      }
    }
}

// ---------------------------------------------------------------------------
// beta[b,h,t] = sigmoid(content[b,t,:] . b_w[h,:] + b_b[h]); one wave per (m,h)
// ---------------------------------------------------------------------------
__global__ __launch_bounds__(256) void beta_kernel(
    const float* __restrict__ content, const float* __restrict__ b_w,
    const float* __restrict__ b_b, float* __restrict__ beta_out) {
  const int gw   = (int)(((size_t)blockIdx.x * blockDim.x + threadIdx.x) >> 5);
  const int lane = threadIdx.x & 31;
  const int h = gw & (H_ - 1);
  const int m = gw >> 4;
  if (m >= B_ * T_) return;
  const float* x = content + (size_t)m * D_;
  const float* w = b_w + (size_t)h * D_;
  float s = 0.0f;
  for (int k = lane; k < D_; k += 32) s += x[k] * w[k];
#pragma unroll
  for (int off = 16; off > 0; off >>= 1) s += __shfl_xor(s, off, 32);
  if (lane == 0) {
    const float z = s + b_b[h];
    const int b = m >> 8, t = m & (T_ - 1);
    beta_out[((size_t)b * H_ + h) * T_ + t] = 1.0f / (1.0f + __expf(-z));
  }
}

// ---------------------------------------------------------------------------
// L2-normalize per (b,t,h) over 64 dims + re-layout (B,T,H*64) -> (B,H,T,64).
// One wave per (m,h); 2 elements per lane.
// ---------------------------------------------------------------------------
__global__ __launch_bounds__(256) void norm_rearrange_kernel(
    const float* __restrict__ in, __bf16* __restrict__ outb,
    float* __restrict__ outf, int do_norm) {
  const int gw   = (int)(((size_t)blockIdx.x * blockDim.x + threadIdx.x) >> 5);
  const int lane = threadIdx.x & 31;
  const int h = gw & (H_ - 1);
  const int m = gw >> 4;
  if (m >= B_ * T_) return;
  const float* src = in + (size_t)m * (H_ * DK_) + h * DK_;
  const float x0 = src[lane];
  const float x1 = src[lane + 32];
  float sc = 1.0f;
  if (do_norm) {
    float ss = x0 * x0 + x1 * x1;
#pragma unroll
    for (int off = 16; off > 0; off >>= 1) ss += __shfl_xor(ss, off, 32);
    const float nrm = sqrtf(ss);
    sc = 1.0f / fmaxf(nrm, 1e-12f);
  }
  const int b = m >> 8, t = m & (T_ - 1);
  const size_t o = ((size_t)(b * H_ + h) * T_ + t) * DK_;
  if (outb) { outb[o + lane] = (__bf16)(x0 * sc); outb[o + lane + 32] = (__bf16)(x1 * sc); }
  if (outf) { outf[o + lane] = x0;                outf[o + lane + 32] = x1; }
}

// ---------------------------------------------------------------------------
// Delta-rule kernel: one workgroup per (b,h). 241 KB dynamic LDS (CDNA5 WGP).
// LDS byte map (dynamic base == 0, no static LDS in this kernel):
//   [0      , 65536 ) Vp    256x64 f32   (v -> V_rhs -> solved Vp)
//   [65536  , 81920 ) Wm    64x64  f32   (W_old -> W_new)
//   [81920  , 82944 ) beta  256    f32
//   [82944  , 115712) Kb    256x64 bf16  (normalized k)
//   [115712 , 246784) Ab    256x256 bf16 (strictly-lower beta_j*k_i.k_j; q later)
// All staging uses CDNA5 GLOBAL_LOAD_ASYNC_TO_LDS_B128 (ASYNCcnt).
// ---------------------------------------------------------------------------
#define LOFF_VP   0u
#define LOFF_WM   65536u
#define LOFF_BETA (65536u + 16384u)
#define LOFF_KB   (65536u + 16384u + 1024u)
#define LOFF_AB   (65536u + 16384u + 1024u + 32768u)
#define DELTA_SMEM (65536 + 16384 + 1024 + 32768 + 131072)

__global__ __launch_bounds__(256) void delta_kernel(
    const __bf16* __restrict__ k_bh, const float* __restrict__ v_bh,
    const __bf16* __restrict__ q_bh, const float* __restrict__ beta_bh,
    const float* __restrict__ W_old, __bf16* __restrict__ merged) {
  extern __shared__ char smem[];
  float*  Vp    = (float*)(smem + LOFF_VP);
  float*  Wm    = (float*)(smem + LOFF_WM);
  float*  betaS = (float*)(smem + LOFF_BETA);
  __bf16* Kb    = (__bf16*)(smem + LOFF_KB);
  __bf16* Ab    = (__bf16*)(smem + LOFF_AB);

  const int tid  = threadIdx.x;
  const int lane = tid & 31;
  const int wave = tid >> 5;
  const int bh   = blockIdx.x;

  // ---- async-stage per-head operands into LDS (16B chunks per lane)
  {
    const char* kg = (const char*)(k_bh    + (size_t)bh * T_ * DK_);
    const char* vg = (const char*)(v_bh    + (size_t)bh * T_ * DV_);
    const char* wg = (const char*)(W_old   + (size_t)bh * DV_ * DK_);
    const char* bg = (const char*)(beta_bh + (size_t)bh * T_);
    for (int i = tid; i < 2048; i += 256)           // Kb: 32 KB
      async_copy_16(LOFF_KB + 16u * i, kg + 16 * i);
    for (int i = tid; i < 4096; i += 256)           // Vp: 64 KB
      async_copy_16(LOFF_VP + 16u * i, vg + 16 * i);
    for (int i = tid; i < 1024; i += 256)           // Wm: 16 KB
      async_copy_16(LOFF_WM + 16u * i, wg + 16 * i);
    if (tid < 64)                                   // beta: 1 KB
      async_copy_16(LOFF_BETA + 16u * tid, bg + 16 * tid);
    wait_async0();
  }
  __syncthreads();

  // ---- (b) V_rhs = v - k * W_old^T    (64 16x16 tiles, 8 per wave)
  for (int it = 0; it < 8; ++it) {
    const int tile = it * 8 + wave;
    const int mt = tile >> 2, nt = tile & 3;
    v8f acc = {};
#pragma unroll
    for (int k0 = 0; k0 < DK_; k0 += 32) {
      v16bf a = load_frag_a(Kb + mt * 16 * DK_ + k0, DK_, lane);
      v16bf b = load_frag_b_nT_f32(Wm + nt * 16 * DK_ + k0, DK_, lane);
      acc = wmma_bf16(a, b, acc);
    }
    const int m0 = mt * 16 + 8 * (lane >> 4);
    const int n0 = nt * 16 + (lane & 15);
#pragma unroll
    for (int r = 0; r < 8; ++r) Vp[(m0 + r) * DV_ + n0] -= acc[r];
  }
  __syncthreads();

  // ---- (c) A[i,j] = beta_j * (k_i . k_j), strictly lower.
  // Lower-or-diagonal 16x16 blocks: 136, round-robin across 8 waves.
  for (int idx = wave; idx < 136; idx += 8) {
    int I = 0;
    while (((I + 1) * (I + 2)) / 2 <= idx) ++I;
    const int J = idx - (I * (I + 1)) / 2;
    v8f acc = {};
#pragma unroll
    for (int k0 = 0; k0 < DK_; k0 += 32) {
      v16bf a = load_frag_a(Kb + I * 16 * DK_ + k0, DK_, lane);
      v16bf b = load_frag_b_nT(Kb + J * 16 * DK_ + k0, DK_, lane);
      acc = wmma_bf16(a, b, acc);
    }
    const int m0 = I * 16 + 8 * (lane >> 4);
    const int n0 = J * 16 + (lane & 15);
#pragma unroll
    for (int r = 0; r < 8; ++r) {
      const int i = m0 + r, j = n0;
      const float val = (j < i) ? betaS[j] * acc[r] : 0.0f;
      Ab[i * 256 + j] = (__bf16)val;
    }
  }
  __syncthreads();

  // ---- (d) blocked forward substitution: (I + A) Vp = V_rhs
  for (int Ib = 0; Ib < 16; ++Ib) {
    if (wave < 4 && Ib > 0) {                    // waves 0..3: N-quarters
      v8f acc = {};
      const int Klen = Ib * 16;
      for (int k0 = 0; k0 < Klen; k0 += 32) {
        int kv = Klen - k0; if (kv > 32) kv = 32;
        v16bf a = load_frag_a(Ab + (size_t)(Ib * 16) * 256 + k0, 256, lane);
        v16bf b = load_frag_b_kn_f32_mask(Vp + k0 * DV_ + wave * 16, DV_, lane, kv);
        acc = wmma_bf16(a, b, acc);
      }
      const int m0 = Ib * 16 + 8 * (lane >> 4);
      const int n0 = wave * 16 + (lane & 15);
#pragma unroll
      for (int r = 0; r < 8; ++r) Vp[(m0 + r) * DV_ + n0] -= acc[r];
    }
    __syncthreads();
    // 16x16 unit-lower diagonal solve: lane v owns column v
    for (int i = 1; i < 16; ++i) {
      if (tid < DV_) {
        const int gi = Ib * 16 + i;
        float sacc = 0.0f;
        for (int j = 0; j < i; ++j)
          sacc += (float)Ab[gi * 256 + Ib * 16 + j] * Vp[(Ib * 16 + j) * DV_ + tid];
        Vp[gi * DV_ + tid] -= sacc;
      }
      __syncthreads();
    }
  }

  // ---- (e) W_new = W_old + (beta .* Vp)^T * k   (16 tiles, 2 per wave)
  for (int it = 0; it < 2; ++it) {
    const int tile = it * 8 + wave;
    const int vt = tile >> 2, dt = tile & 3;
    v8f acc = {};
    for (int k0 = 0; k0 < T_; k0 += 32) {
      v16bf a = load_frag_a_betaVp(Vp, betaS, vt * 16, k0, lane);
      v16bf b = load_frag_b_kn(Kb + k0 * DK_ + dt * 16, DK_, lane);
      acc = wmma_bf16(a, b, acc);
    }
    const int m0 = vt * 16 + 8 * (lane >> 4);
    const int n0 = dt * 16 + (lane & 15);
#pragma unroll
    for (int r = 0; r < 8; ++r) Wm[(m0 + r) * DK_ + n0] += acc[r];
  }
  __syncthreads();

  // ---- (f) read = q * W_new^T  -> merged (B,T,H*DV) bf16  (q async into Ab)
  {
    const char* qg = (const char*)(q_bh + (size_t)bh * T_ * DK_);
    for (int i = tid; i < 2048; i += 256)
      async_copy_16(LOFF_AB + 16u * i, qg + 16 * i);
    wait_async0();
  }
  __syncthreads();
  const int bb = bh >> 4, hh = bh & 15;
  for (int it = 0; it < 8; ++it) {
    const int tile = it * 8 + wave;
    const int mt = tile >> 2, nt = tile & 3;
    v8f acc = {};
#pragma unroll
    for (int k0 = 0; k0 < DK_; k0 += 32) {
      v16bf a = load_frag_a(Ab + mt * 16 * DK_ + k0, DK_, lane);
      v16bf b = load_frag_b_nT_f32(Wm + nt * 16 * DK_ + k0, DK_, lane);
      acc = wmma_bf16(a, b, acc);
    }
    const int m0 = mt * 16 + 8 * (lane >> 4);
    const int n0 = nt * 16 + (lane & 15);
#pragma unroll
    for (int r = 0; r < 8; ++r) {
      const int t = m0 + r;
      merged[(size_t)(bb * T_ + t) * (H_ * DV_) + hh * DV_ + n0] = (__bf16)acc[r];
    }
  }
}

// ---------------------------------------------------------------------------
// Host-side orchestration
// ---------------------------------------------------------------------------
extern "C" void kernel_launch(void* const* d_in, const int* in_sizes, int n_in,
                              void* d_out, int out_size, void* d_ws, size_t ws_size,
                              hipStream_t stream) {
  (void)in_sizes; (void)n_in; (void)out_size; (void)ws_size;
  const float* hidden     = (const float*)d_in[0];
  const float* content    = (const float*)d_in[1];
  const float* W_old      = (const float*)d_in[2];
  const float* q_w        = (const float*)d_in[3];
  const float* o_w        = (const float*)d_in[4];
  const float* k_w        = (const float*)d_in[5];
  const float* v_w        = (const float*)d_in[6];
  const float* b_w        = (const float*)d_in[7];
  const float* b_b        = (const float*)d_in[8];
  const float* read_scale = (const float*)d_in[9];
  float* out = (float*)d_out;
  char*  ws  = (char*)d_ws;

  const size_t MT = (size_t)B_ * T_;   // 4096 rows
  const size_t HD = (size_t)H_ * DK_;  // 1024

  size_t off = 0;
  auto alloc = [&](size_t bytes) {
    size_t o = off; off += (bytes + 255) & ~(size_t)255; return o;
  };
  const size_t o_content_bf = alloc(MT * D_ * 2);
  const size_t o_hidden_bf  = alloc(MT * D_ * 2);
  const size_t o_kw_bf      = alloc(HD * D_ * 2);
  const size_t o_vw_bf      = alloc(HD * D_ * 2);
  const size_t o_qw_bf      = alloc(HD * D_ * 2);
  const size_t o_ow_bf      = alloc((size_t)D_ * HD * 2);
  const size_t o_Kp         = alloc(MT * HD * 4);  // reused as merged bf16
  const size_t o_Vp         = alloc(MT * HD * 4);
  const size_t o_Qp         = alloc(MT * HD * 4);
  const size_t o_kbh        = alloc(MT * HD * 2);
  const size_t o_qbh        = alloc(MT * HD * 2);
  const size_t o_vbh        = alloc(MT * HD * 4);
  const size_t o_beta       = alloc((size_t)B_ * H_ * T_ * 4);

  __bf16* content_bf = (__bf16*)(ws + o_content_bf);
  __bf16* hidden_bf  = (__bf16*)(ws + o_hidden_bf);
  __bf16* kw_bf      = (__bf16*)(ws + o_kw_bf);
  __bf16* vw_bf      = (__bf16*)(ws + o_vw_bf);
  __bf16* qw_bf      = (__bf16*)(ws + o_qw_bf);
  __bf16* ow_bf      = (__bf16*)(ws + o_ow_bf);
  float*  Kp         = (float*)(ws + o_Kp);
  float*  Vpj        = (float*)(ws + o_Vp);
  float*  Qp         = (float*)(ws + o_Qp);
  __bf16* kbh        = (__bf16*)(ws + o_kbh);
  __bf16* qbh        = (__bf16*)(ws + o_qbh);
  float*  vbh        = (float*)(ws + o_vbh);
  float*  betab      = (float*)(ws + o_beta);
  __bf16* merged     = (__bf16*)(ws + o_Kp);      // Kp consumed before delta

  // 1) fp32 -> bf16 conversions
  cvt_f32_bf16_kernel<<<4096, 256, 0, stream>>>(content, content_bf, MT * D_);
  cvt_f32_bf16_kernel<<<4096, 256, 0, stream>>>(hidden,  hidden_bf,  MT * D_);
  cvt_f32_bf16_kernel<<<2048, 256, 0, stream>>>(k_w, kw_bf, HD * D_);
  cvt_f32_bf16_kernel<<<2048, 256, 0, stream>>>(v_w, vw_bf, HD * D_);
  cvt_f32_bf16_kernel<<<2048, 256, 0, stream>>>(q_w, qw_bf, HD * D_);
  cvt_f32_bf16_kernel<<<2048, 256, 0, stream>>>(o_w, ow_bf, (size_t)D_ * HD);

  // 2) projection GEMMs (M=4096, N=1024, K=2048)
  dim3 gproj(HD / 128, MT / 64);
  gemm_bf16_kernel<<<gproj, 256, 0, stream>>>(content_bf, kw_bf, Kp,  nullptr, nullptr,
                                              (int)MT, (int)HD, D_);
  gemm_bf16_kernel<<<gproj, 256, 0, stream>>>(content_bf, vw_bf, Vpj, nullptr, nullptr,
                                              (int)MT, (int)HD, D_);
  gemm_bf16_kernel<<<gproj, 256, 0, stream>>>(hidden_bf,  qw_bf, Qp,  nullptr, nullptr,
                                              (int)MT, (int)HD, D_);

  // 3) beta and normalization / re-layout
  beta_kernel<<<8192, 256, 0, stream>>>(content, b_w, b_b, betab);
  norm_rearrange_kernel<<<8192, 256, 0, stream>>>(Kp,  kbh, nullptr, 1);
  norm_rearrange_kernel<<<8192, 256, 0, stream>>>(Qp,  qbh, nullptr, 1);
  norm_rearrange_kernel<<<8192, 256, 0, stream>>>(Vpj, nullptr, vbh, 0);

  // 4) per-head delta rule (256 workgroups, 241 KB LDS each)
  hipFuncSetAttribute((const void*)delta_kernel,
                      hipFuncAttributeMaxDynamicSharedMemorySize, DELTA_SMEM);
  delta_kernel<<<B_ * H_, 256, DELTA_SMEM, stream>>>(kbh, vbh, qbh, betab, W_old, merged);

  // 5) out = hidden + sigmoid(read_scale) * (merged @ o_w^T)
  dim3 gout(D_ / 128, MT / 64);
  gemm_bf16_kernel<<<gout, 256, 0, stream>>>(merged, ow_bf, out, hidden, read_scale,
                                             (int)MT, D_, (int)HD);
}
